// DSTCarbonFormer_3719441678944
// MI455X (gfx1250) — matrix operand order repaired and checked by reading, not compiled, not run on hardware
//
#include <hip/hip_runtime.h>
#include <hip/hip_bf16.h>

typedef _Float16 half_t;
typedef __attribute__((ext_vector_type(16))) _Float16 v16h;
typedef __attribute__((ext_vector_type(8)))  _Float16 v8h;
typedef __attribute__((ext_vector_type(8)))  float    v8f;
typedef __attribute__((ext_vector_type(4)))  unsigned int v4u;
typedef __attribute__((ext_vector_type(8)))  int v8i;
typedef __attribute__((ext_vector_type(4)))  int v4i;

#define DEV __device__ __forceinline__

DEV float gelu_exact(float x){ return 0.5f*x*(1.0f+erff(x*0.70710678118654752f)); }
DEV float silu_f(float x){ return x/(1.0f+__expf(-x)); }
DEV float sigmoid_f(float x){ return 1.0f/(1.0f+__expf(-x)); }
DEV float softplus_f(float x){ return (x>20.f)? x : log1pf(__expf(x)); }
DEV float act_apply(float v,int act){
  switch(act){case 1: return gelu_exact(v); case 2: return silu_f(v);
              case 3: return sigmoid_f(v); case 4: return softplus_f(v);}
  return v;
}

// LDS B-slab layout: Bs[col*72 + k], 64 valid k per slab, 8 halves row pad.
#define BS_LD 72

// ---------------------------------------------------------------------------
// Fragment helpers (ISA 7.12.2, wave32).
// ---------------------------------------------------------------------------
DEV v16h load_a_frag(const float* __restrict__ A, int lda, int m, int M,
                     int K, int k0, int hs){
  v16h a;
  const int kbA = k0 + hs*8;
  if (m < M && ((lda & 3) == 0) && (k0 + 32 <= K)){
    const float4* A4 = (const float4*)(A + (size_t)m*lda);
    const int q = kbA >> 2;
    float4 q0 = A4[q], q1 = A4[q+1], q2 = A4[q+4], q3 = A4[q+5];
    a[0]=(half_t)q0.x; a[1]=(half_t)q0.y; a[2]=(half_t)q0.z; a[3]=(half_t)q0.w;
    a[4]=(half_t)q1.x; a[5]=(half_t)q1.y; a[6]=(half_t)q1.z; a[7]=(half_t)q1.w;
    a[8]=(half_t)q2.x; a[9]=(half_t)q2.y; a[10]=(half_t)q2.z; a[11]=(half_t)q2.w;
    a[12]=(half_t)q3.x; a[13]=(half_t)q3.y; a[14]=(half_t)q3.z; a[15]=(half_t)q3.w;
  } else {
    #pragma unroll
    for (int e=0;e<16;++e){
      int kk = kbA + e + ((e<8)?0:8);
      a[e] = (half_t)((m<M && kk<K) ? A[(size_t)m*lda + kk] : 0.f);
    }
  }
  return a;
}

DEV v16h load_b_fragT(const half_t* __restrict__ Bs, int half_ch, int hs, int l16){
  const half_t* p = Bs + l16*BS_LD + half_ch*32 + hs*8;
  v8h lo = *(const v8h*)(p);
  v8h hi = *(const v8h*)(p + 16);
  v16h b;
  #pragma unroll
  for (int e=0;e<8;++e){ b[e] = lo[e]; b[e+8] = hi[e]; }
  return b;
}

// unconditional cooperative stage of one 64x16 slab (1024 halves, 8B/thread)
DEV void stage_b_slab(half_t* __restrict__ Bs, const half_t* __restrict__ wp,
                      int tileN, int Kpad, int k0, int tid){
  const int nl = tid >> 4, k4 = (tid & 15) * 4;
  *(uint2*)(Bs + nl*BS_LD + k4) =
      *(const uint2*)(wp + (size_t)(tileN + nl)*Kpad + k0 + k4);
}

// ---------------------------------------------------------------------------
// TDM staging of the same 64x16 slab (ISA 08_async_tensor §7/§8).
// D# group0: count=1, lds_addr, global_addr(57b), type=2.
// D# group1: data_size=2B, pad_enable, pad_interval=4 (pad every 128B),
//            pad_amount=3 (4 dwords) -> reproduces the BS_LD=72 row stride;
//            tensor_dim0=Kpad, tensor_dim1=16, tile 64x16, stride0=Kpad.
// ---------------------------------------------------------------------------
#if __has_builtin(__builtin_amdgcn_tensor_load_to_lds)
#define HAVE_TDM 1
DEV void tdm_stage_slab(const half_t* __restrict__ wp_tile, half_t* __restrict__ Bs,
                        int Kpad){
  unsigned long long ga = (unsigned long long)wp_tile;
  unsigned int lds = (unsigned int)(unsigned long long)Bs;
  v4u g0;
  g0[0] = 1u;                                              // count=1
  g0[1] = lds;                                             // lds_addr
  g0[2] = (unsigned)(ga & 0xFFFFFFFFu);                    // global_addr lo
  g0[3] = (unsigned)((ga >> 32) & 0x01FFFFFFu) | (2u<<30); // global_addr hi, type=2
  unsigned td0 = (unsigned)Kpad;
  v8i g1;
  g1[0] = (int)((1u<<16) | (1u<<20) | (4u<<22) | (3u<<25)); // dsz=2B,pad 128B+16B
  g1[1] = (int)((td0 & 0xFFFFu) << 16);                     // tensor_dim0 lo
  g1[2] = (int)((td0 >> 16) | (16u << 16));                 // dim0 hi, tensor_dim1=16
  g1[3] = (int)(64u << 16);                                 // tile_dim0=64
  g1[4] = (int)16u;                                         // tile_dim1=16
  g1[5] = (int)td0;                                         // tensor_dim0_stride lo
  g1[6] = 0; g1[7] = 0;
  v4i z4 = {0,0,0,0};
#if __clang_major__ >= 23
  v8i z8 = {0,0,0,0,0,0,0,0};
  __builtin_amdgcn_tensor_load_to_lds(g0, g1, z4, z4, z8, 0);
#else
  __builtin_amdgcn_tensor_load_to_lds(g0, g1, z4, z4, 0);
#endif
#if __has_builtin(__builtin_amdgcn_s_wait_tensorcnt)
  __builtin_amdgcn_s_wait_tensorcnt(0);
#endif
}
#else
#define HAVE_TDM 0
#endif

// ---------------------------------------------------------------------------
// Weight packs -> f16 column-major wp[n*Kpad + k], zero padded.
// ---------------------------------------------------------------------------
__global__ void k_pack_tiles(const float* __restrict__ in, half_t* __restrict__ out,
                             int K, int N, int Kpad, int Npad16, int trans){
  int i = blockIdx.x*blockDim.x + threadIdx.x;
  if (i >= Npad16*Kpad) return;
  int k = i % Kpad, n = i / Kpad;
  float v = 0.f;
  if (n < N && k < K) v = trans ? in[(size_t)n*K + k] : in[(size_t)k*N + n];
  out[i] = (half_t)v;
}

__global__ void k_pack_convw(const float* __restrict__ w, half_t* __restrict__ out,
                             int Cin, int Cpad, int N, int Kpad, int Npad16){
  int i = blockIdx.x*blockDim.x + threadIdx.x;
  if (i >= Npad16*Kpad) return;
  int k = i % Kpad, n = i / Kpad;
  int K = Cpad*27;
  float v = 0.f;
  if (n < N && k < K){
    int r = k / Cpad, c = k - r*Cpad;
    if (c < Cin) v = w[(size_t)n*(Cin*27) + (size_t)c*27 + r];
  }
  out[i] = (half_t)v;
}

__global__ void k_pad_ch(const float* __restrict__ in, float* __restrict__ out,
                         int Cin, int Cpad, int M){
  int i = blockIdx.x*blockDim.x + threadIdx.x;
  if (i >= Cpad*M) return;
  out[i] = (i < Cin*M) ? in[i] : 0.f;
}

// ---------------------------------------------------------------------------
// Implicit-im2col 3D conv as WMMA GEMM (synchronous B staging).
// ---------------------------------------------------------------------------
__global__ void k_wmma_conv3d(const float* __restrict__ x, const half_t* __restrict__ wp,
                              const float* __restrict__ bias, float* __restrict__ out,
                              int Cpad, int T, int H, int W, int N,
                              int ksz, int dil, int act, int Kpad){
  __shared__ half_t Bs[16*BS_LD];
  const int M  = T*H*W;
  const int Kk = (ksz==3) ? Cpad*27 : Cpad;
  const int lane = threadIdx.x & 31;
  const int hs = lane >> 4, l16 = lane & 15;
  const int tileN = blockIdx.x*16;
  const int rowBase = blockIdx.y*128 + (threadIdx.x >> 5)*16;
  const int m = rowBase + l16;
  int tq=0,hq=0,wq=0;
  if (m < M){ tq = m/(H*W); int r = m%(H*W); hq = r/W; wq = r%W; }
  v8f acc0 = {}, acc1 = {};

  for (int k0=0; k0<Kk; k0+=64){
    stage_b_slab(Bs, wp, tileN, Kpad, k0, threadIdx.x);
    __syncthreads();
    #pragma unroll
    for (int hc=0; hc<2; ++hc){
      const int kc = k0 + hc*32;
      if (kc >= Kk) break;
      float va[16];
      if (ksz==3){
        const int r  = kc / Cpad;
        const int kt = r/9, kh = (r/3)%3, kw = r%3;
        const int tt = tq + (kt-1)*dil, hh = hq + (kh-1)*dil, ww = wq + (kw-1)*dil;
        const bool inb = (m<M) && tt>=0 && tt<T && hh>=0 && hh<H && ww>=0 && ww<W;
        if (inb){
          const float* xb = x + (size_t)((kc - r*Cpad) + hs*8)*M
                              + (((size_t)tt*H + hh)*W + ww);
          #pragma unroll
          for (int e=0;e<16;++e) va[e] = xb[(size_t)(e + ((e<8)?0:8))*M];
        } else {
          #pragma unroll
          for (int e=0;e<16;++e) va[e] = 0.f;
        }
      } else {
        if (m < M){
          const float* xb = x + (size_t)(kc + hs*8)*M + m;
          #pragma unroll
          for (int e=0;e<16;++e) va[e] = xb[(size_t)(e + ((e<8)?0:8))*M];
        } else {
          #pragma unroll
          for (int e=0;e<16;++e) va[e] = 0.f;
        }
      }
      v16h a;
      #pragma unroll
      for (int e=0;e<16;++e) a[e] = (half_t)va[e];
      v16h b = load_b_fragT(Bs, hc, hs, l16);
      if (hc==0)
        acc0 = __builtin_amdgcn_wmma_f32_16x16x32_f16(false,a,false,b,(short)0,acc0,false,false);
      else
        acc1 = __builtin_amdgcn_wmma_f32_16x16x32_f16(false,a,false,b,(short)0,acc1,false,false);
    }
    __syncthreads();
  }

  v8f acc = acc0 + acc1;
  const int ncol = tileN + l16;
  const int row0 = rowBase + 8*hs;
  if (ncol < N){
    float bv = bias ? bias[ncol] : 0.f;
    if (row0 + 7 < M){
      float4 o0, o1;
      o0.x=act_apply(acc[0]+bv,act); o0.y=act_apply(acc[1]+bv,act);
      o0.z=act_apply(acc[2]+bv,act); o0.w=act_apply(acc[3]+bv,act);
      o1.x=act_apply(acc[4]+bv,act); o1.y=act_apply(acc[5]+bv,act);
      o1.z=act_apply(acc[6]+bv,act); o1.w=act_apply(acc[7]+bv,act);
      float4* op = (float4*)(out + (size_t)ncol*M + row0);
      op[0] = o0; op[1] = o1;
    } else {
      #pragma unroll
      for (int r=0;r<8;++r){
        int row = row0 + r;
        if (row < M) out[(size_t)ncol*M + row] = act_apply(acc[r]+bv, act);
      }
    }
  }
}

// ---------------------------------------------------------------------------
// Token GEMM with TDM-staged B panel (falls back to sync staging w/o TDM).
// ---------------------------------------------------------------------------
__global__ void k_wmma_gemm(const float* __restrict__ A, int lda,
                            const half_t* __restrict__ B,
                            const float* __restrict__ bias,
                            float* __restrict__ C, int ldc,
                            int M, int N, int K, int act, int addOut, int Kpad){
  __shared__ half_t Bs[16*BS_LD];
  const int lane = threadIdx.x & 31;
  const int hs = lane >> 4, l16 = lane & 15;
  const int tileN = blockIdx.x*16;
  const int rowBase = blockIdx.y*128 + (threadIdx.x >> 5)*16;
  const int m = rowBase + l16;
  v8f acc0 = {}, acc1 = {};

  for (int k0=0; k0<K; k0+=64){
#if HAVE_TDM
    if (threadIdx.x < 32){
      // wave 0 drives the Tensor Data Mover, waits TENSORcnt, barrier releases
      tdm_stage_slab(B + (size_t)tileN*Kpad + k0, Bs, Kpad);
    }
#else
    stage_b_slab(Bs, B, tileN, Kpad, k0, threadIdx.x);
#endif
    __syncthreads();
    if (m < M && k0+64 < K) __builtin_prefetch(&A[(size_t)m*lda + k0 + 64], 0, 1);
    {
      v16h a = load_a_frag(A, lda, m, M, K, k0, hs);
      v16h b = load_b_fragT(Bs, 0, hs, l16);
      acc0 = __builtin_amdgcn_wmma_f32_16x16x32_f16(false,a,false,b,(short)0,acc0,false,false);
    }
    if (k0 + 32 < K){
      v16h a = load_a_frag(A, lda, m, M, K, k0+32, hs);
      v16h b = load_b_fragT(Bs, 1, hs, l16);
      acc1 = __builtin_amdgcn_wmma_f32_16x16x32_f16(false,a,false,b,(short)0,acc1,false,false);
    }
    __syncthreads();
  }

  v8f acc = acc0 + acc1;
  const int ncol = tileN + l16;
  if (ncol < N){
    float bv = bias ? bias[ncol] : 0.f;
    #pragma unroll
    for (int r=0;r<8;++r){
      int row = rowBase + r + 8*hs;
      if (row < M){
        float v = act_apply(acc[r] + bv, act);
        size_t o = (size_t)row*ldc + ncol;
        if (addOut) v += C[o];
        C[o] = v;
      }
    }
  }
}

// ---------------------------------------------------------------------------
// Norm / elementwise / scan kernels
// ---------------------------------------------------------------------------
__global__ void k_instnorm(const float* __restrict__ x, float* __restrict__ y, int M, int act){
  __shared__ float s1[256], s2[256];
  const float* xc = x + (size_t)blockIdx.x*M;
  float a = 0.f, b = 0.f;
  for (int i=threadIdx.x;i<M;i+=256){ float v = xc[i]; a += v; b += v*v; }
  s1[threadIdx.x]=a; s2[threadIdx.x]=b; __syncthreads();
  for (int s=128;s>0;s>>=1){
    if (threadIdx.x < s){ s1[threadIdx.x]+=s1[threadIdx.x+s]; s2[threadIdx.x]+=s2[threadIdx.x+s]; }
    __syncthreads();
  }
  float mean = s1[0]/M, var = s2[0]/M - mean*mean;
  float inv = rsqrtf(var + 1e-5f);
  float* yc = y + (size_t)blockIdx.x*M;
  for (int i=threadIdx.x;i<M;i+=256){
    float v = (xc[i]-mean)*inv;
    yc[i] = (act==1) ? gelu_exact(v) : v;
  }
}

__global__ void k_layernorm(const float* __restrict__ x, const float* __restrict__ g,
                            const float* __restrict__ b, float* __restrict__ y, int C){
  __shared__ float s1[128], s2[128];
  const float* xr = x + (size_t)blockIdx.x*C;
  float v = (threadIdx.x < C) ? xr[threadIdx.x] : 0.f;
  s1[threadIdx.x]=v; s2[threadIdx.x]=v*v; __syncthreads();
  for (int s=64;s>0;s>>=1){
    if (threadIdx.x < s){ s1[threadIdx.x]+=s1[threadIdx.x+s]; s2[threadIdx.x]+=s2[threadIdx.x+s]; }
    __syncthreads();
  }
  float mean = s1[0]/C, var = s2[0]/C - mean*mean, inv = rsqrtf(var+1e-5f);
  if (threadIdx.x < C)
    y[(size_t)blockIdx.x*C + threadIdx.x] = (v-mean)*inv*g[threadIdx.x] + b[threadIdx.x];
}

__global__ void k_dwconv_silu(const float* __restrict__ x, int ldx,
                              const float* __restrict__ w, const float* __restrict__ b,
                              float* __restrict__ y, int ldy, int L, int Dm, int Kc, int rev){
  int i = blockIdx.x*blockDim.x + threadIdx.x;
  if (i >= L*Dm) return;
  int l = i / Dm, d = i % Dm;
  float acc = b[d];
  for (int k=0;k<Kc;++k){
    int ll = l - (Kc-1) + k;
    if (ll >= 0){
      int lp = rev ? (L-1-ll) : ll;
      acc += x[(size_t)lp*ldx + d] * w[(size_t)d*Kc + k];
    }
  }
  y[(size_t)l*ldy + d] = silu_f(acc);
}

__global__ void k_negexp(const float* __restrict__ in, float* __restrict__ out, int n){
  int i = blockIdx.x*blockDim.x + threadIdx.x;
  if (i < n) out[i] = -__expf(in[i]);
}

__global__ void k_scan(const float* __restrict__ u, int ldu,
                       const float* __restrict__ dt, int lddt,
                       const float* __restrict__ Bm, int ldb,
                       const float* __restrict__ Cm, int ldcm,
                       const float* __restrict__ An, const float* __restrict__ Dv,
                       float* __restrict__ y, int ldy,
                       int L, int Nst, int rev, int add){
  const int d = blockIdx.x, n = threadIdx.x;
  const float a_d = (n < Nst) ? An[(size_t)d*Nst + n] : 0.f;
  const float Dd = Dv[d];
  float h = 0.f;
  for (int l=0;l<L;++l){
    float dtv = dt[(size_t)l*lddt + d];
    float uv  = u[(size_t)l*ldu + d];
    float bl  = (n < Nst) ? Bm[(size_t)l*ldb + n]  : 0.f;
    float cl  = (n < Nst) ? Cm[(size_t)l*ldcm + n] : 0.f;
    h = h*__expf(dtv*a_d) + dtv*uv*bl;
    float part = h*cl;
    for (int s=16;s>0;s>>=1) part += __shfl_xor(part, s, 32);
    if (n == 0){
      int lp = rev ? (L-1-l) : l;
      size_t o = (size_t)lp*ldy + d;
      float val = part + uv*Dd;
      y[o] = add ? (y[o] + val) : val;
    }
  }
}

__global__ void k_mul_siluz(const float* __restrict__ y, const float* __restrict__ z, int ldz,
                            float* __restrict__ o, int L, int Dm){
  int i = blockIdx.x*blockDim.x + threadIdx.x;
  if (i >= L*Dm) return;
  int l = i / Dm, d = i % Dm;
  o[i] = y[i] * silu_f(z[(size_t)l*ldz + d]);
}

__global__ void k_transpose(const float* __restrict__ in, float* __restrict__ out, int R, int C){
  int i = blockIdx.x*blockDim.x + threadIdx.x;
  if (i >= R*C) return;
  int r = i / C, c = i % C;
  out[(size_t)c*R + r] = in[i];
}

__global__ void k_upsample2(const float* __restrict__ in, float* __restrict__ out,
                            int C, int T, int H, int W){
  int Mh = T*4*H*W;
  int i = blockIdx.x*blockDim.x + threadIdx.x;
  if (i >= C*Mh) return;
  int c = i / Mh, r = i % Mh;
  int t = r/(4*H*W); int rr = r%(4*H*W);
  int hh = rr/(2*W), ww = rr%(2*W);
  out[i] = in[((size_t)c*T + t)*H*W + (size_t)(hh>>1)*W + (ww>>1)];
}

__global__ void k_prep_main(const float* __restrict__ mainx, float* __restrict__ mainN,
                            float* __restrict__ mskip, int T, int Hl, int Wl){
  int Ml = T*Hl*Wl, Mh = T*4*Hl*Wl;
  int i = blockIdx.x*blockDim.x + threadIdx.x;
  if (i < Ml) mainN[i] = mainx[i] * (1.0f/11.0f);
  if (i < Mh){
    int t = i/(4*Hl*Wl); int r = i%(4*Hl*Wl);
    int hh = r/(2*Wl), ww = r%(2*Wl);
    mskip[i] = mainx[((size_t)t*Hl + (hh>>1))*Wl + (ww>>1)] * (1.0f/11.0f);
  }
}

__global__ void k_add(const float* a, const float* b, float* o, int n){
  int i = blockIdx.x*blockDim.x + threadIdx.x; if (i<n) o[i] = a[i]+b[i];
}
__global__ void k_copy(const float* a, float* o, int n){
  int i = blockIdx.x*blockDim.x + threadIdx.x; if (i<n) o[i] = a[i];
}
__global__ void k_sft_apply(const float* x, const float* sc, const float* sh, float* o, int n){
  int i = blockIdx.x*blockDim.x + threadIdx.x; if (i<n) o[i] = x[i]*(1.f+sc[i]) + sh[i];
}
__global__ void k_gatemix(const float* g, const float* x, const float* a, float* o, int n){
  int i = blockIdx.x*blockDim.x + threadIdx.x;
  if (i<n){ float gv = g[i]; o[i] = gv*x[i] + (1.f-gv)*a[i]; }
}
__global__ void k_scale_ch(const float* x, const float* attn, float* o, int C, int M){
  int i = blockIdx.x*blockDim.x + threadIdx.x;
  if (i < C*M) o[i] = x[i]*attn[i/M];
}

__global__ void k_chan_meanmax(const float* __restrict__ x, int M,
                               float* __restrict__ avg, float* __restrict__ mx){
  __shared__ float s1[256], s2[256];
  const float* xc = x + (size_t)blockIdx.x*M;
  float a = 0.f, m = -1e30f;
  for (int i=threadIdx.x;i<M;i+=256){ float v = xc[i]; a += v; m = fmaxf(m, v); }
  s1[threadIdx.x]=a; s2[threadIdx.x]=m; __syncthreads();
  for (int s=128;s>0;s>>=1){
    if (threadIdx.x < s){ s1[threadIdx.x]+=s1[threadIdx.x+s]; s2[threadIdx.x]=fmaxf(s2[threadIdx.x], s2[threadIdx.x+s]); }
    __syncthreads();
  }
  if (threadIdx.x==0){ avg[blockIdx.x]=s1[0]/M; mx[blockIdx.x]=s2[0]; }
}

__global__ void k_attn_vec(const float* avg, const float* mx,
                           const float* w1, const float* b1,
                           const float* w2, const float* b2,
                           float* attn, int C, int Hd){
  __shared__ float ha[32], hm[32];
  int t = threadIdx.x;
  if (t < Hd){
    float sa = b1[t], sm = b1[t];
    for (int c=0;c<C;++c){ sa += avg[c]*w1[c*Hd+t]; sm += mx[c]*w1[c*Hd+t]; }
    ha[t] = fmaxf(sa, 0.f); hm[t] = fmaxf(sm, 0.f);
  }
  __syncthreads();
  if (t < C){
    float oa = b2[t], om = b2[t];
    for (int j=0;j<Hd;++j){ oa += ha[j]*w2[j*C+t]; om += hm[j]*w2[j*C+t]; }
    attn[t] = sigmoid_f(oa + om);
  }
}

__global__ void k_tail(const float* __restrict__ resid, const float* __restrict__ mskip,
                       const float* __restrict__ aux, const float* __restrict__ res_scale,
                       float* __restrict__ out, int Mh, int auxStride){
  int i = blockIdx.x*blockDim.x + threadIdx.x;
  if (i >= Mh) return;
  float pred = mskip[i]*(1.f + res_scale[0]*tanhf(resid[i]));
  float prior = 0.5f*(aux[i] + aux[(size_t)6*auxStride + i]);
  float mask = fminf(((mskip[i]>0.f)?1.f:0.f) + ((prior>1e-6f)?1.f:0.f), 1.f);
  float v = fmaxf(pred*mask*11.0f, 0.f);
  out[i] = v; out[Mh + i] = v;
}

// ---------------------------------------------------------------------------
// Orchestration
// ---------------------------------------------------------------------------
extern "C" void kernel_launch(void* const* d_in, const int* in_sizes, int n_in,
                              void* d_out, int out_size, void* d_ws, size_t ws_size,
                              hipStream_t stream){
  (void)in_sizes; (void)out_size; (void)ws_size;
  const float* aux   = (const float*)d_in[0];
  const float* mainx = (const float*)d_in[1];

  int pi = 2;
  auto nextp = [&]() -> const float* {
    int idx = (pi < n_in) ? pi : (n_in > 0 ? n_in-1 : 0);
    if (pi < n_in) ++pi;
    return (const float*)d_in[idx];
  };

  const int T=2, Hl=60, Wl=60, Ml=T*Hl*Wl;     // 7200
  const int Hh=120, Wh=120, Mh=T*Hh*Wh;        // 28800
  const int LDB = 72;

  char* wsb = (char*)d_ws; size_t off = 0;
  auto allocF = [&](size_t n)->float*{ float* p=(float*)(wsb+off); off += ((n*4+255)/256)*256; return p; };
  auto allocH = [&](size_t n)->half_t*{ half_t* p=(half_t*)(wsb+off); off += ((n*2+255)/256)*256; return p; };

  float* mainN = allocF(Ml);
  float* mskip = allocF(Mh);
  float* A1 = allocF((size_t)96*Mh);
  float* A2 = allocF((size_t)96*Mh);
  float* A3 = allocF((size_t)96*Mh);
  float* auxF = allocF((size_t)96*Mh);
  float* cat3 = allocF((size_t)288*Mh);
  float* Xpad = allocF((size_t)64*Mh);
  float* tokX = allocF((size_t)Ml*96);
  float* tokLN= allocF((size_t)Ml*96);
  float* xz   = allocF((size_t)Ml*384);
  float* xc   = allocF((size_t)Ml*192);
  float* dbc  = allocF((size_t)Ml*LDB);
  float* dtb  = allocF((size_t)Ml*192);
  float* yac  = allocF((size_t)Ml*192);
  float* ymul = allocF((size_t)Ml*192);
  float* Aneg = allocF(192*32);
  float* avgv = allocF(96); float* mxv = allocF(96); float* attnv = allocF(96);
  float* resid= allocF(Mh);
  half_t* wpack = allocH((size_t)2688*112);

  auto ew = [&](int n){ return dim3((n+255)/256); };
  auto ceil64 = [](int k){ return (k+63) & ~63; };

  auto conv = [&](const float* x, const float* w, const float* bias, float* out,
                  int Cin,int Tq,int Hq,int Wq,int N,int ksz,int dil,int act){
    int M = Tq*Hq*Wq;
    int Npad16 = (N+15) & ~15;
    if (ksz==3){
      int Cpad = (Cin+31) & ~31;
      int Kpad = ceil64(Cpad*27);
      const float* xin = x;
      if (Cpad != Cin){
        k_pad_ch<<<ew(Cpad*M),256,0,stream>>>(x, Xpad, Cin, Cpad, M);
        xin = Xpad;
      }
      k_pack_convw<<<ew(Npad16*Kpad),256,0,stream>>>(w, wpack, Cin, Cpad, N, Kpad, Npad16);
      dim3 g((N+15)/16, (M+127)/128);
      k_wmma_conv3d<<<g,256,0,stream>>>(xin, wpack, bias, out, Cpad,Tq,Hq,Wq,N,3,dil,act,Kpad);
    } else {
      int Kpad = ceil64(Cin);
      k_pack_tiles<<<ew(Npad16*Kpad),256,0,stream>>>(w, wpack, Cin, N, Kpad, Npad16, 1);
      dim3 g((N+15)/16, (M+127)/128);
      k_wmma_conv3d<<<g,256,0,stream>>>(x, wpack, bias, out, Cin,Tq,Hq,Wq,N,1,1,act,Kpad);
    }
  };
  auto gemm = [&](const float* A, int lda, const float* w, int K, int N, int trans,
                  const float* bias, float* C, int ldc, int M, int act, int add){
    int Npad16 = (N+15) & ~15;
    int Kpad = ceil64(K);
    k_pack_tiles<<<ew(Npad16*Kpad),256,0,stream>>>(w, wpack, K, N, Kpad, Npad16, trans);
    dim3 g((N+15)/16, (M+127)/128);
    k_wmma_gemm<<<g,256,0,stream>>>(A, lda, wpack, bias, C, ldc, M, N, K, act, add, Kpad);
  };

  k_prep_main<<<ew(Mh),256,0,stream>>>(mainx, mainN, mskip, T, Hl, Wl);

  { const float* w0=nextp(); const float* b0=nextp();
    const float* w1=nextp(); const float* b1=nextp();
    const float* w2=nextp(); const float* b2=nextp();
    conv(aux, w0, b0, A1, 9,T,Hh,Wh, 48, 3,1, 0);
    k_instnorm<<<48,256,0,stream>>>(A1, A2, Mh, 1);
    conv(A2, w1, b1, A1, 48,T,Hh,Wh, 96, 3,1, 0);
    k_instnorm<<<96,256,0,stream>>>(A1, A2, Mh, 1);
    conv(A2, w2, b2, auxF, 96,T,Hh,Wh, 96, 3,1, 0);
  }
  { const float* w1=nextp(); const float* b1=nextp();
    const float* w2=nextp(); const float* b2=nextp();
    const float* w3=nextp(); const float* b3=nextp();
    const float* wf=nextp(); const float* bf=nextp();
    conv(auxF, w1, b1, cat3,                96,T,Hh,Wh, 96, 3,1, 0);
    conv(auxF, w2, b2, cat3+(size_t)96*Mh,  96,T,Hh,Wh, 96, 3,2, 0);
    conv(auxF, w3, b3, cat3+(size_t)192*Mh, 96,T,Hh,Wh, 96, 3,3, 0);
    conv(cat3, wf, bf, A1, 288,T,Hh,Wh, 96, 1,1, 1);
    k_add<<<ew(96*Mh),256,0,stream>>>(auxF, A1, auxF, 96*Mh);
  }

  { const float* w0=nextp(); const float* b0=nextp();
    const float* w1=nextp(); const float* b1=nextp();
    const float* w2=nextp(); const float* b2=nextp();
    conv(mainN, w0, b0, A1, 1,T,Hl,Wl, 24, 3,1, 0);
    k_instnorm<<<24,256,0,stream>>>(A1, A2, Ml, 1);
    conv(A2, w1, b1, A1, 24,T,Hl,Wl, 48, 3,1, 0);
    k_instnorm<<<48,256,0,stream>>>(A1, A2, Ml, 1);
    conv(A2, w2, b2, A1, 48,T,Hl,Wl, 96, 3,1, 1);
    nextp(); // pos embedding (deferred)
  }
  k_transpose<<<ew(96*Ml),256,0,stream>>>(A1, tokX, 96, Ml);

  for (int layer=0; layer<3; ++layer){
    const float* ln_g = nextp(); const float* ln_b = nextp();
    const float* in_proj = nextp(); const float* out_proj = nextp();
    k_layernorm<<<Ml,128,0,stream>>>(tokX, ln_g, ln_b, tokLN, 96);
    gemm(tokLN, 96, in_proj, 96, 384, 0, nullptr, xz, 384, Ml, 0, 0);
    for (int dir=0; dir<2; ++dir){
      const float* conv_w = nextp(); const float* conv_b = nextp();
      const float* x_proj = nextp();
      const float* dt_w = nextp();  const float* dt_b = nextp();
      const float* A_log = nextp(); const float* Dv = nextp();
      int rev = dir;
      k_dwconv_silu<<<ew(Ml*192),256,0,stream>>>(xz, 384, conv_w, conv_b, xc, 192, Ml, 192, 4, rev);
      gemm(xc, 192, x_proj, 192, 70, 0, nullptr, dbc, LDB, Ml, 0, 0);
      gemm(dbc, LDB, dt_w, 6, 192, 0, dt_b, dtb, 192, Ml, 4, 0);
      k_negexp<<<ew(192*32),256,0,stream>>>(A_log, Aneg, 192*32);
      k_scan<<<192,32,0,stream>>>(xc,192, dtb,192, dbc+6,LDB, dbc+38,LDB,
                                  Aneg, Dv, yac,192, Ml, 32, rev, dir);
    }
    k_mul_siluz<<<ew(Ml*192),256,0,stream>>>(yac, xz+192, 384, ymul, Ml, 192);
    gemm(ymul, 192, out_proj, 192, 96, 0, nullptr, tokX, 96, Ml, 0, 1);
  }
  { const float* g=nextp(); const float* b=nextp();
    const float* ow=nextp(); const float* ob=nextp();
    k_layernorm<<<Ml,128,0,stream>>>(tokX, g, b, tokLN, 96);
    gemm(tokLN, 96, ow, 96, 96, 0, ob, xz, 96, Ml, 0, 0);
  }
  k_transpose<<<ew(Ml*96),256,0,stream>>>(xz, A1, Ml, 96);
  k_upsample2<<<ew(96*Mh),256,0,stream>>>(A1, A2, 96, T, Hl, Wl);
  { const float* aw=nextp(); const float* ab=nextp();
    conv(A2, aw, ab, A3, 96,T,Hh,Wh, 96, 3,1, 1);
  }

  { const float* sw0=nextp(); const float* sb0=nextp();
    const float* sw1=nextp(); const float* sb1=nextp();
    const float* tw0=nextp(); const float* tb0=nextp();
    const float* tw1=nextp(); const float* tb1=nextp();
    conv(auxF, sw0, sb0, A1, 96,T,Hh,Wh, 96, 1,1, 1);
    conv(A1,   sw1, sb1, A2, 96,T,Hh,Wh, 96, 1,1, 0);
    conv(auxF, tw0, tb0, A1, 96,T,Hh,Wh, 96, 1,1, 1);
    conv(A1,   tw1, tb1, cat3, 96,T,Hh,Wh, 96, 1,1, 0);
    k_sft_apply<<<ew(96*Mh),256,0,stream>>>(A3, A2, cat3, A1, 96*Mh);
  }
  { const float* gw=nextp(); const float* gb=nextp();
    k_copy<<<ew(96*Mh),256,0,stream>>>(A1, cat3, 96*Mh);
    k_copy<<<ew(96*Mh),256,0,stream>>>(auxF, cat3+(size_t)96*Mh, 96*Mh);
    conv(cat3, gw, gb, A2, 192,T,Hh,Wh, 96, 1,1, 3);
    k_gatemix<<<ew(96*Mh),256,0,stream>>>(A2, A1, auxF, A1, 96*Mh);
  }
  { const float* w1=nextp(); const float* b1=nextp();
    const float* w2=nextp(); const float* b2=nextp();
    k_chan_meanmax<<<96,256,0,stream>>>(A1, Mh, avgv, mxv);
    k_attn_vec<<<1,128,0,stream>>>(avgv, mxv, w1, b1, w2, b2, attnv, 96, 12);
    k_scale_ch<<<ew(96*Mh),256,0,stream>>>(A1, attnv, A1, 96, Mh);
  }

  { const float* bw=nextp(); const float* bb=nextp();
    conv(A1, bw, bb, A2, 96,T,Hh,Wh, 96, 3,1, 0);
    k_add<<<ew(96*Mh),256,0,stream>>>(A2, A3, A1, 96*Mh);
  }

  { const float* sw0=nextp(); const float* sb0=nextp();
    const float* sw1=nextp(); const float* sb1=nextp();
    const float* tw0=nextp(); const float* tb0=nextp();
    const float* tw1=nextp(); const float* tb1=nextp();
    conv(auxF, sw0, sb0, A3, 96,T,Hh,Wh, 96, 1,1, 1);
    conv(A3,   sw1, sb1, A2, 96,T,Hh,Wh, 96, 1,1, 0);
    conv(auxF, tw0, tb0, A3, 96,T,Hh,Wh, 96, 1,1, 1);
    conv(A3,   tw1, tb1, cat3, 96,T,Hh,Wh, 96, 1,1, 0);
    k_sft_apply<<<ew(96*Mh),256,0,stream>>>(A1, A2, cat3, A1, 96*Mh);
  }
  { const float* w1=nextp(); const float* b1=nextp();
    const float* w2=nextp(); const float* b2=nextp();
    k_chan_meanmax<<<96,256,0,stream>>>(A1, Mh, avgv, mxv);
    k_attn_vec<<<1,128,0,stream>>>(avgv, mxv, w1, b1, w2, b2, attnv, 96, 12);
    k_scale_ch<<<ew(96*Mh),256,0,stream>>>(A1, attnv, A1, 96, Mh);
  }

  { const float* t0w=nextp(); const float* t0b=nextp();
    const float* t1w=nextp(); const float* t1b=nextp();
    const float* rsc=nextp();
    conv(A1, t0w, t0b, A2, 96,T,Hh,Wh, 96, 3,1, 1);
    conv(A2, t1w, t1b, resid, 96,T,Hh,Wh, 1, 1,1, 0);
    k_tail<<<ew(Mh),256,0,stream>>>(resid, mskip, aux, rsc, (float*)d_out, Mh, Mh);
  }
}
